// PointcloudEncoder_1769526526404
// MI455X (gfx1250) — matrix operand order repaired and checked by reading, not compiled, not run on hardware
//
#include <hip/hip_runtime.h>
#include <math.h>

// ---------------- problem constants ----------------
#define Bc    4
#define Nn    8192
#define Gg    512
#define Kk    32
#define Dd    768
#define Hh    12
#define NDEPTH 4
#define ENCC  256
#define MLPD  3072
#define HDd   64
#define Ss    513          // G + 1
#define MT    (Bc*Ss)      // 2052
#define MPAD  2080         // padded row count for activation buffers
#define SPAD  544          // 513 keys padded to 17*32

// ---------------- WMMA helpers (gfx1250 wave32) ----------------
typedef __attribute__((ext_vector_type(16))) _Float16 v16h;
typedef __attribute__((ext_vector_type(8)))  _Float16 v8h;
typedef __attribute__((ext_vector_type(8)))  float    v8f;

__device__ __forceinline__ v8f zero_v8f() {
  v8f z;
  #pragma unroll
  for (int i = 0; i < 8; ++i) z[i] = 0.f;
  return z;
}
__device__ __forceinline__ v16h zero_v16h() {
  v16h z;
  #pragma unroll
  for (int i = 0; i < 16; ++i) z[i] = (_Float16)0.f;
  return z;
}

// A fragment: 16x32 f16, row-major source with leading dim lda (halves).
// lane&15 = row m; chunk0 at k0 + (lane>=16 ? 16:0), chunk1 at +8.
__device__ __forceinline__ v16h load_afrag(const _Float16* A, int lda, int m0, int k0, int lane) {
  const _Float16* p = A + (size_t)(m0 + (lane & 15)) * lda + k0 + ((lane & 16) ? 16 : 0);
  v8h lo = *(const v8h*)p;
  v8h hi = *(const v8h*)(p + 8);
  v16h r;
  #pragma unroll
  for (int i = 0; i < 8; ++i) { r[i] = lo[i]; r[8 + i] = hi[i]; }
  return r;
}

// B fragment: 32x16 (KxN). Weights stored as Wp[N][K] f16 so each lane reads
// 16 contiguous halves: row n0+(lane&15), k offset k0 + (lane>=16 ? 16:0).
__device__ __forceinline__ v16h load_bfrag(const _Float16* Wp, int ldk, int n0, int k0, int lane) {
  const _Float16* p = Wp + (size_t)(n0 + (lane & 15)) * ldk + k0 + ((lane & 16) ? 16 : 0);
  return *(const v16h*)p;
}

__device__ __forceinline__ v8f wmma16(v16h a, v16h b, v8f c) {
  return __builtin_amdgcn_wmma_f32_16x16x32_f16(false, a, false, b, (short)0, c, false, false);
}

__device__ __forceinline__ float gelu_exact(float x) {
  return 0.5f * x * (1.f + erff(x * 0.70710678118654752f));
}

// ---------------- FPS (serial argmax chain) ----------------
__global__ __launch_bounds__(1024) void fps_kernel(const float* __restrict__ pts,
                                                   int* __restrict__ fidx,
                                                   float* __restrict__ center) {
  int b = blockIdx.x;
  const float* P = pts + (size_t)b * Nn * 3;
  __shared__ float redv[1024];
  __shared__ int   redi[1024];
  __shared__ int   scur;
  int tid = threadIdx.x;
  float dmin[8];
  #pragma unroll
  for (int j = 0; j < 8; ++j) dmin[j] = 1e10f;
  if (tid == 0) {
    scur = 0;
    fidx[b * Gg] = 0;
    center[(size_t)(b * Gg) * 3 + 0] = P[0];
    center[(size_t)(b * Gg) * 3 + 1] = P[1];
    center[(size_t)(b * Gg) * 3 + 2] = P[2];
  }
  __syncthreads();
  for (int it = 1; it < Gg; ++it) {
    int cur = scur;
    float cx = P[cur * 3 + 0], cy = P[cur * 3 + 1], cz = P[cur * 3 + 2];
    float best = -1.f; int bi = 0;
    #pragma unroll
    for (int j = 0; j < 8; ++j) {
      int i = tid + j * 1024;
      float dx = P[i * 3 + 0] - cx, dy = P[i * 3 + 1] - cy, dz = P[i * 3 + 2] - cz;
      float d = dx * dx + dy * dy + dz * dz;
      if (d < dmin[j]) dmin[j] = d;
      if (dmin[j] > best) { best = dmin[j]; bi = i; }
    }
    redv[tid] = best; redi[tid] = bi;
    __syncthreads();
    for (int s = 512; s > 0; s >>= 1) {
      if (tid < s) {
        float v2 = redv[tid + s]; int i2 = redi[tid + s];
        if (v2 > redv[tid] || (v2 == redv[tid] && i2 < redi[tid])) { redv[tid] = v2; redi[tid] = i2; }
      }
      __syncthreads();
    }
    if (tid == 0) {
      int nx = redi[0];
      scur = nx;
      fidx[b * Gg + it] = nx;
      center[(size_t)(b * Gg + it) * 3 + 0] = P[nx * 3 + 0];
      center[(size_t)(b * Gg + it) * 3 + 1] = P[nx * 3 + 1];
      center[(size_t)(b * Gg + it) * 3 + 2] = P[nx * 3 + 2];
    }
    __syncthreads();
  }
}

// ---------------- KNN: 32 smallest distances per center ----------------
__global__ __launch_bounds__(256) void knn_kernel(const float* __restrict__ pts,
                                                  const float* __restrict__ center,
                                                  int* __restrict__ knn) {
  int bg = blockIdx.x;
  int b = bg / Gg;
  const float* P = pts + (size_t)b * Nn * 3;
  float cx = center[bg * 3 + 0], cy = center[bg * 3 + 1], cz = center[bg * 3 + 2];
  __shared__ float sd[Nn];
  __shared__ float redv[256];
  __shared__ int   redi[256];
  int tid = threadIdx.x;
  for (int j = 0; j < 32; ++j) {
    int i = tid + j * 256;
    float dx = P[i * 3 + 0] - cx, dy = P[i * 3 + 1] - cy, dz = P[i * 3 + 2] - cz;
    sd[i] = dx * dx + dy * dy + dz * dz;
  }
  __syncthreads();
  for (int kk = 0; kk < Kk; ++kk) {
    float best = 3.0e38f; int bi = 0;
    for (int j = 0; j < 32; ++j) {
      int i = tid + j * 256;
      float v = sd[i];
      if (v < best || (v == best && i < bi)) { best = v; bi = i; }
    }
    redv[tid] = best; redi[tid] = bi;
    __syncthreads();
    for (int s = 128; s > 0; s >>= 1) {
      if (tid < s) {
        float v2 = redv[tid + s]; int i2 = redi[tid + s];
        if (v2 < redv[tid] || (v2 == redv[tid] && i2 < redi[tid])) { redv[tid] = v2; redi[tid] = i2; }
      }
      __syncthreads();
    }
    if (tid == 0) { knn[(size_t)bg * Kk + kk] = redi[0]; sd[redi[0]] = 3.4e38f; }
    __syncthreads();
  }
}

// ---------------- weight conversion kernels ----------------
__global__ void cvt_kernel(const float* __restrict__ in, _Float16* __restrict__ out, int n) {
  int i = blockIdx.x * 256 + threadIdx.x;
  if (i < n) out[i] = (_Float16)in[i];
}

// in [R][C] f32 -> out [C][R] f16 (tiled, coalesced)
__global__ void transpose_cvt(const float* __restrict__ in, _Float16* __restrict__ out, int R, int C) {
  __shared__ float tile[32][33];
  int c0 = blockIdx.x * 32, r0 = blockIdx.y * 32;
  int tx = threadIdx.x, ty = threadIdx.y;
  for (int j = 0; j < 32; j += 8) {
    int r = r0 + ty + j, c = c0 + tx;
    tile[ty + j][tx] = (r < R && c < C) ? in[(size_t)r * C + c] : 0.f;
  }
  __syncthreads();
  for (int j = 0; j < 32; j += 8) {
    int c = c0 + ty + j, r = r0 + tx;
    if (c < C && r < R) out[(size_t)c * R + r] = (_Float16)tile[tx][ty + j];
  }
}

// ---------------- fused per-group PointNet encoder ----------------
__global__ __launch_bounds__(256) void encoder_kernel(
    const float* __restrict__ pts, const float* __restrict__ colors,
    const float* __restrict__ center, const int* __restrict__ knn,
    const float* __restrict__ w1a, const float* __restrict__ b1a,
    const float* __restrict__ g1, const float* __restrict__ bb1,
    const float* __restrict__ m1, const float* __restrict__ v1,
    const _Float16* __restrict__ w1b16, const float* __restrict__ b1b,
    const _Float16* __restrict__ w2a16, const float* __restrict__ b2a,
    const float* __restrict__ g2, const float* __restrict__ bb2,
    const float* __restrict__ m2, const float* __restrict__ v2,
    const _Float16* __restrict__ w2b16, const float* __restrict__ b2b,
    _Float16* __restrict__ tok16) {
  __shared__ __align__(32) char buf[65536];
  _Float16* f1   = (_Float16*)buf;                   // [32][128] f16 (8KB)   region X
  float*    feat = (float*)(buf + 8192);             // [32][6]   f32         region X
  _Float16* f2   = (_Float16*)(buf + 32768);         // [32][256] f16 (16KB)  region Y
  float*    fg   = (float*)(buf + 32768 + 16384);    // [256]     f32         region Y
  _Float16* fcat = (_Float16*)buf;                   // [32][512] f16 (32KB)  region X
  _Float16* f3   = (_Float16*)(buf + 32768);         // [32][512] f16 (32KB)  region Y
  float*    f4   = (float*)buf;                      // [32][256] f32 (32KB)  region X

  int bg = blockIdx.x;
  int b = bg / Gg;
  int tid = threadIdx.x, lane = tid & 31, wid = tid >> 5;

  // gather neighborhood features
  if (tid < Kk) {
    int idx = knn[(size_t)bg * Kk + tid];
    const float* p = pts    + ((size_t)b * Nn + idx) * 3;
    const float* c = colors + ((size_t)b * Nn + idx) * 3;
    feat[tid * 6 + 0] = p[0] - center[bg * 3 + 0];
    feat[tid * 6 + 1] = p[1] - center[bg * 3 + 1];
    feat[tid * 6 + 2] = p[2] - center[bg * 3 + 2];
    feat[tid * 6 + 3] = c[0];
    feat[tid * 6 + 4] = c[1];
    feat[tid * 6 + 5] = c[2];
  }
  __syncthreads();

  // layer 1a: 6 -> 128, BN + ReLU (K-dim too small for WMMA; VALU)
  for (int i = tid; i < 32 * 128; i += 256) {
    int r = i >> 7, cc = i & 127;
    float s = b1a[cc];
    #pragma unroll
    for (int j = 0; j < 6; ++j) s += feat[r * 6 + j] * w1a[cc * 6 + j];
    s = (s - m1[cc]) * rsqrtf(v1[cc] + 1e-5f) * g1[cc] + bb1[cc];
    s = fmaxf(s, 0.f);
    f1[r * 128 + cc] = (_Float16)s;
  }
  __syncthreads();

  // layer 1b: M=32, K=128, N=256 (WMMA) -> f2 (+bias)
  {
    int mt = wid >> 2, ntb = (wid & 3) * 4;
    v8f acc[4];
    #pragma unroll
    for (int j = 0; j < 4; ++j) acc[j] = zero_v8f();
    for (int ks = 0; ks < 4; ++ks) {
      v16h a = load_afrag(f1, 128, mt * 16, ks * 32, lane);
      #pragma unroll
      for (int j = 0; j < 4; ++j)
        acc[j] = wmma16(a, load_bfrag(w1b16, 128, (ntb + j) * 16, ks * 32, lane), acc[j]);
    }
    int nl = lane & 15, ms = (lane & 16) ? 8 : 0;
    #pragma unroll
    for (int j = 0; j < 4; ++j) {
      int n = (ntb + j) * 16 + nl;
      float bv = b1b[n];
      #pragma unroll
      for (int r = 0; r < 8; ++r) {
        int m = mt * 16 + ms + r;
        f2[m * 256 + n] = (_Float16)(acc[j][r] + bv);
      }
    }
  }
  __syncthreads();

  // global (per-group) max pool of f2
  {
    float mx = -3e38f;
    for (int r = 0; r < 32; ++r) mx = fmaxf(mx, (float)f2[r * 256 + tid]);
    fg[tid] = mx;
  }
  __syncthreads();

  // concat [fg, f2] -> fcat [32][512]
  for (int i = tid; i < 32 * 512; i += 256) {
    int r = i >> 9, cc = i & 511;
    float v = (cc < 256) ? fg[cc] : (float)f2[r * 256 + (cc - 256)];
    fcat[i] = (_Float16)v;
  }
  __syncthreads();

  // layer 2a: M=32, K=512, N=512 (WMMA), BN + ReLU -> f3
  {
    int mt = wid >> 2, ntb = (wid & 3) * 8;
    v8f acc[8];
    #pragma unroll
    for (int j = 0; j < 8; ++j) acc[j] = zero_v8f();
    for (int ks = 0; ks < 16; ++ks) {
      v16h a = load_afrag(fcat, 512, mt * 16, ks * 32, lane);
      #pragma unroll
      for (int j = 0; j < 8; ++j)
        acc[j] = wmma16(a, load_bfrag(w2a16, 512, (ntb + j) * 16, ks * 32, lane), acc[j]);
    }
    int nl = lane & 15, ms = (lane & 16) ? 8 : 0;
    #pragma unroll
    for (int j = 0; j < 8; ++j) {
      int n = (ntb + j) * 16 + nl;
      float bv = b2a[n];
      float sc = rsqrtf(v2[n] + 1e-5f) * g2[n];
      float mm = m2[n], bb = bb2[n];
      #pragma unroll
      for (int r = 0; r < 8; ++r) {
        int m = mt * 16 + ms + r;
        float v = (acc[j][r] + bv - mm) * sc + bb;
        f3[m * 512 + n] = (_Float16)fmaxf(v, 0.f);
      }
    }
  }
  __syncthreads();

  // layer 2b: M=32, K=512, N=256 (WMMA) -> f4 (+bias, f32)
  {
    int mt = wid >> 2, ntb = (wid & 3) * 4;
    v8f acc[4];
    #pragma unroll
    for (int j = 0; j < 4; ++j) acc[j] = zero_v8f();
    for (int ks = 0; ks < 16; ++ks) {
      v16h a = load_afrag(f3, 512, mt * 16, ks * 32, lane);
      #pragma unroll
      for (int j = 0; j < 4; ++j)
        acc[j] = wmma16(a, load_bfrag(w2b16, 512, (ntb + j) * 16, ks * 32, lane), acc[j]);
    }
    int nl = lane & 15, ms = (lane & 16) ? 8 : 0;
    #pragma unroll
    for (int j = 0; j < 4; ++j) {
      int n = (ntb + j) * 16 + nl;
      float bv = b2b[n];
      #pragma unroll
      for (int r = 0; r < 8; ++r) {
        int m = mt * 16 + ms + r;
        f4[m * 256 + n] = acc[j][r] + bv;
      }
    }
  }
  __syncthreads();

  // token = max over K points
  {
    float mx = -3e38f;
    for (int r = 0; r < 32; ++r) mx = fmaxf(mx, f4[r * 256 + tid]);
    tok16[(size_t)bg * 256 + tid] = (_Float16)mx;
  }
}

// ---------------- generic 128x128 WMMA GEMM with fused epilogues ----------
#define EP_BIAS      0
#define EP_BIAS_RES  1
#define EP_GELU      2

__global__ __launch_bounds__(256) void gemm128(
    const _Float16* __restrict__ A,   // [M][Kd] f16 row-major
    const _Float16* __restrict__ Wp,  // [N][Kd] f16 (pre-transposed weight)
    const float* __restrict__ bias,   // [N] or null
    const float* __restrict__ resid,  // [M][N] f32 or null (EP_BIAS_RES)
    float* __restrict__ Cf,           // optional f32 out
    _Float16* __restrict__ Ch,        // optional f16 out
    int M, int Kd, int N, int mode) {
  __shared__ __align__(32) _Float16 smA[128 * 32];
  __shared__ __align__(32) _Float16 smB[128 * 32];
  int tid = threadIdx.x, lane = tid & 31, wid = tid >> 5;
  int m0 = blockIdx.y * 128, n0 = blockIdx.x * 128;
  v8f acc[8];
  #pragma unroll
  for (int i = 0; i < 8; ++i) acc[i] = zero_v8f();

  for (int k0 = 0; k0 < Kd; k0 += 32) {
    int row = tid >> 1;
    int off = (tid & 1) * 16;
    int mg = m0 + row;
    v16h va = zero_v16h();
    if (mg < M) va = *(const v16h*)(A + (size_t)mg * Kd + k0 + off);
    *(v16h*)(smA + row * 32 + off) = va;
    v16h vb = *(const v16h*)(Wp + (size_t)(n0 + row) * Kd + k0 + off);
    *(v16h*)(smB + row * 32 + off) = vb;
    __syncthreads();

    v16h a = load_afrag(smA, 32, wid * 16, 0, lane);
    #pragma unroll
    for (int nt = 0; nt < 8; ++nt)
      acc[nt] = wmma16(a, load_bfrag(smB, 32, nt * 16, 0, lane), acc[nt]);
    __syncthreads();
  }

  int nl = lane & 15, ms = (lane & 16) ? 8 : 0;
  #pragma unroll
  for (int nt = 0; nt < 8; ++nt) {
    int n = n0 + nt * 16 + nl;
    float bv = bias ? bias[n] : 0.f;
    #pragma unroll
    for (int r = 0; r < 8; ++r) {
      int m = m0 + wid * 16 + ms + r;
      if (m < M) {
        float v = acc[nt][r] + bv;
        if (mode == EP_GELU)      v = gelu_exact(v);
        else if (mode == EP_BIAS_RES) v += resid[(size_t)m * N + n];
        if (Cf) Cf[(size_t)m * N + n] = v;
        if (Ch) Ch[(size_t)m * N + n] = (_Float16)v;
      }
    }
  }
}

// ---------------- misc small kernels ----------------
__global__ void pos1_kernel(const float* __restrict__ center, const float* __restrict__ w,
                            const float* __restrict__ bias, _Float16* __restrict__ out) {
  int i = blockIdx.x * 256 + threadIdx.x;
  if (i >= Bc * Gg * 128) return;
  int row = i >> 7, c = i & 127;
  float s = bias[c] + center[row * 3 + 0] * w[c * 3 + 0]
                    + center[row * 3 + 1] * w[c * 3 + 1]
                    + center[row * 3 + 2] * w[c * 3 + 2];
  out[i] = (_Float16)gelu_exact(s);
}

__global__ void assemble_xs(const float* __restrict__ tok768, const float* __restrict__ pos768,
                            const float* __restrict__ clsT, const float* __restrict__ clsP,
                            float* __restrict__ xs) {
  int i = blockIdx.x * 256 + threadIdx.x;
  if (i >= MT * Dd) return;
  int m = i / Dd, n = i - m * Dd;
  int b = m / Ss, r = m - b * Ss;
  float v;
  if (r == 0) v = clsT[n] + clsP[n];
  else {
    int mg = b * Gg + (r - 1);
    v = tok768[(size_t)mg * Dd + n] + pos768[(size_t)mg * Dd + n];
  }
  xs[i] = v;
}

__global__ __launch_bounds__(256) void ln_kernel(const float* __restrict__ x,
                                                 const float* __restrict__ g,
                                                 const float* __restrict__ b,
                                                 _Float16* __restrict__ out) {
  int m = blockIdx.x;
  const float* row = x + (size_t)m * Dd;
  __shared__ float rs[256], rs2[256];
  int tid = threadIdx.x;
  float s = 0.f, s2 = 0.f;
  for (int j = tid; j < Dd; j += 256) { float v = row[j]; s += v; s2 += v * v; }
  rs[tid] = s; rs2[tid] = s2;
  __syncthreads();
  for (int st = 128; st > 0; st >>= 1) {
    if (tid < st) { rs[tid] += rs[tid + st]; rs2[tid] += rs2[tid + st]; }
    __syncthreads();
  }
  float mu = rs[0] * (1.f / Dd);
  float var = rs2[0] * (1.f / Dd) - mu * mu;
  float inv = rsqrtf(var + 1e-5f);
  for (int j = tid; j < Dd; j += 256)
    out[(size_t)m * Dd + j] = (_Float16)((row[j] - mu) * inv * g[j] + b[j]);
}

// V transpose + zero pad: vt[b][h][d][t] over SPAD keys
__global__ void vt_kernel(const _Float16* __restrict__ qkv16, _Float16* __restrict__ vt) {
  int i = blockIdx.x * 256 + threadIdx.x;
  int total = Bc * Hh * HDd * SPAD;
  if (i >= total) return;
  int t = i % SPAD;
  int rest = i / SPAD;
  int d = rest % HDd; rest /= HDd;
  int h = rest % Hh;
  int b = rest / Hh;
  _Float16 v = (_Float16)0.f;
  if (t < Ss) v = qkv16[((size_t)(b * Ss + t)) * (3 * Dd) + 2 * Dd + h * HDd + d];
  vt[i] = v;
}

// ---------------- fused attention (16 query rows per block) --------------
__global__ __launch_bounds__(256) void attn_kernel(const _Float16* __restrict__ qkv16,
                                                   const _Float16* __restrict__ vt,
                                                   _Float16* __restrict__ attO16) {
  int qt = blockIdx.x, h = blockIdx.y, b = blockIdx.z;
  int q0 = qt * 16;
  __shared__ __align__(32) _Float16 smQ[16 * 64];
  __shared__ float smS[16 * SPAD];
  __shared__ float red[256];
  int tid = threadIdx.x, lane = tid & 31, wid = tid >> 5;

  for (int i = tid; i < 16 * 64; i += 256) {
    int r = i >> 6, d = i & 63;
    int mrow = q0 + r;
    smQ[i] = (mrow < Ss) ? qkv16[((size_t)(b * Ss + mrow)) * (3 * Dd) + h * HDd + d]
                         : (_Float16)0.f;
  }
  __syncthreads();

  // scores = Q @ K^T  (WMMA; key tiles spread over 8 waves)
  v16h a0 = load_afrag(smQ, 64, 0, 0, lane);
  v16h a1 = load_afrag(smQ, 64, 0, 32, lane);
  for (int nt = wid; nt < SPAD / 16; nt += 8) {
    int t0 = nt * 16;
    v8f acc = zero_v8f();
    int t = t0 + (lane & 15);
    int trow = (t < Ss) ? t : 0;
    const _Float16* kbase = qkv16 + ((size_t)(b * Ss + trow)) * (3 * Dd) + Dd + h * HDd;
    acc = wmma16(a0, *(const v16h*)(kbase + 0  + ((lane & 16) ? 16 : 0)), acc);
    acc = wmma16(a1, *(const v16h*)(kbase + 32 + ((lane & 16) ? 16 : 0)), acc);
    int nl = lane & 15, ms = (lane & 16) ? 8 : 0;
    #pragma unroll
    for (int r = 0; r < 8; ++r) smS[(ms + r) * SPAD + t0 + nl] = acc[r];
  }
  __syncthreads();

  // masked softmax (scale = 1/sqrt(64))
  int row = tid >> 4, sub = tid & 15;
  float vreg[SPAD / 16];
  float lmax = -3.0e38f;
  #pragma unroll
  for (int j = 0; j < SPAD / 16; ++j) {
    int c = sub + j * 16;
    float v = smS[row * SPAD + c] * 0.125f;
    if (c >= Ss) v = -3.0e38f;
    vreg[j] = v;
    lmax = fmaxf(lmax, v);
  }
  red[tid] = lmax;
  __syncthreads();
  float rmax = -3.0e38f;
  for (int s2 = 0; s2 < 16; ++s2) rmax = fmaxf(rmax, red[row * 16 + s2]);
  float lsum = 0.f;
  #pragma unroll
  for (int j = 0; j < SPAD / 16; ++j) { float e = __expf(vreg[j] - rmax); vreg[j] = e; lsum += e; }
  __syncthreads();
  red[tid] = lsum;
  __syncthreads();
  float tsum = 0.f;
  for (int s2 = 0; s2 < 16; ++s2) tsum += red[row * 16 + s2];
  float inv = 1.f / tsum;
  #pragma unroll
  for (int j = 0; j < SPAD / 16; ++j) smS[row * SPAD + sub + j * 16] = vreg[j] * inv;
  __syncthreads();

  // O = P @ V (WMMA; 4 output d-tiles on waves 0..3)
  if (wid < 4) {
    int d0 = wid * 16;
    v8f acc = zero_v8f();
    for (int ks = 0; ks < SPAD / 32; ++ks) {
      v16h a;
      int m = lane & 15;
      int kb = ks * 32 + ((lane & 16) ? 16 : 0);
      #pragma unroll
      for (int e = 0; e < 8; ++e) {
        a[e]     = (_Float16)smS[m * SPAD + kb + e];
        a[8 + e] = (_Float16)smS[m * SPAD + kb + 8 + e];
      }
      int d = d0 + (lane & 15);
      int kb2 = ks * 32 + ((lane & 16) ? 16 : 0);
      const _Float16* p = vt + (((size_t)(b * Hh + h)) * HDd + d) * SPAD + kb2;
      acc = wmma16(a, *(const v16h*)p, acc);
    }
    int nl = lane & 15, ms = (lane & 16) ? 8 : 0;
    #pragma unroll
    for (int r = 0; r < 8; ++r) {
      int mrow = q0 + ms + r;
      if (mrow < Ss)
        attO16[((size_t)(b * Ss + mrow)) * Dd + h * HDd + d0 + nl] = (_Float16)acc[r];
    }
  }
}

// ---------------- host launcher ----------------
extern "C" void kernel_launch(void* const* d_in, const int* in_sizes, int n_in,
                              void* d_out, int out_size, void* d_ws, size_t ws_size,
                              hipStream_t stream) {
  (void)in_sizes; (void)n_in; (void)out_size; (void)ws_size;
  const float* pts      = (const float*)d_in[0];
  const float* colors   = (const float*)d_in[1];
  const float* enc_w1a  = (const float*)d_in[2];
  const float* enc_b1a  = (const float*)d_in[3];
  const float* bn1_g    = (const float*)d_in[4];
  const float* bn1_b    = (const float*)d_in[5];
  const float* bn1_m    = (const float*)d_in[6];
  const float* bn1_v    = (const float*)d_in[7];
  const float* enc_w1b  = (const float*)d_in[8];
  const float* enc_b1b  = (const float*)d_in[9];
  const float* enc_w2a  = (const float*)d_in[10];
  const float* enc_b2a  = (const float*)d_in[11];
  const float* bn2_g    = (const float*)d_in[12];
  const float* bn2_b    = (const float*)d_in[13];
  const float* bn2_m    = (const float*)d_in[14];
  const float* bn2_v    = (const float*)d_in[15];
  const float* enc_w2b  = (const float*)d_in[16];
  const float* enc_b2b  = (const float*)d_in[17];
  const float* e2t_w    = (const float*)d_in[18];
  const float* e2t_b    = (const float*)d_in[19];
  const float* cls_tok  = (const float*)d_in[20];
  const float* cls_pos  = (const float*)d_in[21];
  const float* pe_w1    = (const float*)d_in[22];
  const float* pe_b1    = (const float*)d_in[23];
  const float* pe_w2    = (const float*)d_in[24];
  const float* pe_b2    = (const float*)d_in[25];
  const float* ln1_g    = (const float*)d_in[26];
  const float* ln1_b    = (const float*)d_in[27];
  const float* qkv_w    = (const float*)d_in[28];
  const float* qkv_b    = (const float*)d_in[29];
  const float* proj_w   = (const float*)d_in[30];
  const float* proj_b   = (const float*)d_in[31];
  const float* ln2_g    = (const float*)d_in[32];
  const float* ln2_b    = (const float*)d_in[33];
  const float* mlp_w1   = (const float*)d_in[34];
  const float* mlp_b1   = (const float*)d_in[35];
  const float* mlp_w2   = (const float*)d_in[36];
  const float* mlp_b2   = (const float*)d_in[37];
  const float* fcn_g    = (const float*)d_in[38];
  const float* fcn_b    = (const float*)d_in[39];
  const float* t2e_w    = (const float*)d_in[40];
  const float* t2e_b    = (const float*)d_in[41];

  // ---- workspace bump allocator (total ~112 MB) ----
  char* wbase = (char*)d_ws;
  size_t off = 0;
  auto alloc = [&](size_t bytes) -> char* {
    char* p = wbase + off;
    off += (bytes + 255) & ~(size_t)255;
    return p;
  };
  int*       fidx    = (int*)alloc((size_t)Bc * Gg * 4);
  float*     center  = (float*)alloc((size_t)Bc * Gg * 3 * 4);
  int*       knn     = (int*)alloc((size_t)Bc * Gg * Kk * 4);
  _Float16*  tok16   = (_Float16*)alloc((size_t)Bc * Gg * 256 * 2);
  _Float16*  pos1_16 = (_Float16*)alloc((size_t)Bc * Gg * 128 * 2);
  float*     tok768  = (float*)alloc((size_t)Bc * Gg * Dd * 4);
  float*     pos768  = (float*)alloc((size_t)Bc * Gg * Dd * 4);
  float*     xs      = (float*)alloc((size_t)MPAD * Dd * 4);
  _Float16*  xn16    = (_Float16*)alloc((size_t)MPAD * Dd * 2);
  _Float16*  qkv16   = (_Float16*)alloc((size_t)MPAD * 3 * Dd * 2);
  _Float16*  vt16    = (_Float16*)alloc((size_t)Bc * Hh * HDd * SPAD * 2);
  _Float16*  attO16  = (_Float16*)alloc((size_t)MPAD * Dd * 2);
  _Float16*  h1_16   = (_Float16*)alloc((size_t)MPAD * MLPD * 2);
  _Float16*  w1b16   = (_Float16*)alloc((size_t)256 * 128 * 2);
  _Float16*  w2a16   = (_Float16*)alloc((size_t)512 * 512 * 2);
  _Float16*  w2b16   = (_Float16*)alloc((size_t)256 * 512 * 2);
  _Float16*  e2t16   = (_Float16*)alloc((size_t)Dd * 256 * 2);
  _Float16*  pe2_16  = (_Float16*)alloc((size_t)Dd * 128 * 2);
  _Float16*  t2e16   = (_Float16*)alloc((size_t)Dd * Dd * 2);
  _Float16*  qkvT16  = (_Float16*)alloc((size_t)NDEPTH * 3 * Dd * Dd * 2);
  _Float16*  projT16 = (_Float16*)alloc((size_t)NDEPTH * Dd * Dd * 2);
  _Float16*  mlp1T16 = (_Float16*)alloc((size_t)NDEPTH * MLPD * Dd * 2);
  _Float16*  mlp2T16 = (_Float16*)alloc((size_t)NDEPTH * Dd * MLPD * 2);

  // ---- geometry ----
  fps_kernel<<<Bc, 1024, 0, stream>>>(pts, fidx, center);
  knn_kernel<<<Bc * Gg, 256, 0, stream>>>(pts, center, knn);

  // ---- weight conversions (f32 -> f16, [N][K] layout) ----
  auto cvt = [&](const float* in, _Float16* out, int n) {
    cvt_kernel<<<(n + 255) / 256, 256, 0, stream>>>(in, out, n);
  };
  cvt(enc_w1b, w1b16, 256 * 128);
  cvt(enc_w2a, w2a16, 512 * 512);
  cvt(enc_w2b, w2b16, 256 * 512);
  cvt(e2t_w,   e2t16, Dd * 256);
  cvt(pe_w2,   pe2_16, Dd * 128);
  cvt(t2e_w,   t2e16, Dd * Dd);
  for (int l = 0; l < NDEPTH; ++l) {
    transpose_cvt<<<dim3((3 * Dd) / 32, Dd / 32), dim3(32, 8), 0, stream>>>(
        qkv_w + (size_t)l * Dd * 3 * Dd, qkvT16 + (size_t)l * 3 * Dd * Dd, Dd, 3 * Dd);
    transpose_cvt<<<dim3(Dd / 32, Dd / 32), dim3(32, 8), 0, stream>>>(
        proj_w + (size_t)l * Dd * Dd, projT16 + (size_t)l * Dd * Dd, Dd, Dd);
    transpose_cvt<<<dim3(MLPD / 32, Dd / 32), dim3(32, 8), 0, stream>>>(
        mlp_w1 + (size_t)l * Dd * MLPD, mlp1T16 + (size_t)l * MLPD * Dd, Dd, MLPD);
    transpose_cvt<<<dim3(Dd / 32, MLPD / 32), dim3(32, 8), 0, stream>>>(
        mlp_w2 + (size_t)l * MLPD * Dd, mlp2T16 + (size_t)l * Dd * MLPD, MLPD, Dd);
  }

  // ---- fused group encoder -> tokens ----
  encoder_kernel<<<Bc * Gg, 256, 0, stream>>>(
      pts, colors, center, knn,
      enc_w1a, enc_b1a, bn1_g, bn1_b, bn1_m, bn1_v,
      w1b16, enc_b1b,
      w2a16, enc_b2a, bn2_g, bn2_b, bn2_m, bn2_v,
      w2b16, enc_b2b, tok16);

  // ---- token/pos projection ----
  pos1_kernel<<<(Bc * Gg * 128 + 255) / 256, 256, 0, stream>>>(center, pe_w1, pe_b1, pos1_16);
  gemm128<<<dim3(Dd / 128, (Bc * Gg + 127) / 128), 256, 0, stream>>>(
      pos1_16, pe2_16, pe_b2, nullptr, pos768, nullptr, Bc * Gg, 128, Dd, EP_BIAS);
  gemm128<<<dim3(Dd / 128, (Bc * Gg + 127) / 128), 256, 0, stream>>>(
      tok16, e2t16, e2t_b, nullptr, tok768, nullptr, Bc * Gg, 256, Dd, EP_BIAS);
  assemble_xs<<<(MT * Dd + 255) / 256, 256, 0, stream>>>(tok768, pos768, cls_tok, cls_pos, xs);

  // ---- transformer blocks ----
  int gy = (MT + 127) / 128;
  for (int l = 0; l < NDEPTH; ++l) {
    ln_kernel<<<MT, 256, 0, stream>>>(xs, ln1_g + l * Dd, ln1_b + l * Dd, xn16);
    gemm128<<<dim3((3 * Dd) / 128, gy), 256, 0, stream>>>(
        xn16, qkvT16 + (size_t)l * 3 * Dd * Dd, qkv_b + l * 3 * Dd, nullptr,
        nullptr, qkv16, MT, Dd, 3 * Dd, EP_BIAS);
    vt_kernel<<<(Bc * Hh * HDd * SPAD + 255) / 256, 256, 0, stream>>>(qkv16, vt16);
    attn_kernel<<<dim3((Ss + 15) / 16, Hh, Bc), 256, 0, stream>>>(qkv16, vt16, attO16);
    gemm128<<<dim3(Dd / 128, gy), 256, 0, stream>>>(
        attO16, projT16 + (size_t)l * Dd * Dd, proj_b + l * Dd, xs,
        xs, nullptr, MT, Dd, Dd, EP_BIAS_RES);
    ln_kernel<<<MT, 256, 0, stream>>>(xs, ln2_g + l * Dd, ln2_b + l * Dd, xn16);
    gemm128<<<dim3(MLPD / 128, gy), 256, 0, stream>>>(
        xn16, mlp1T16 + (size_t)l * MLPD * Dd, mlp_b1 + l * MLPD, nullptr,
        nullptr, h1_16, MT, Dd, MLPD, EP_GELU);
    gemm128<<<dim3(Dd / 128, gy), 256, 0, stream>>>(
        h1_16, mlp2T16 + (size_t)l * Dd * MLPD, mlp_b2 + l * Dd, xs,
        xs, nullptr, MT, MLPD, Dd, EP_BIAS_RES);
  }

  // ---- final LN + projection to output ----
  ln_kernel<<<MT, 256, 0, stream>>>(xs, fcn_g, fcn_b, xn16);
  gemm128<<<dim3(Dd / 128, gy), 256, 0, stream>>>(
      xn16, t2e16, t2e_b, nullptr, (float*)d_out, nullptr, MT, Dd, Dd, EP_BIAS);
}
